// NTXentLoss_83107617178413
// MI455X (gfx1250) — matrix operand order repaired and checked by reading, not compiled, not run on hardware
//
#include <hip/hip_runtime.h>
#include <hip/hip_bf16.h>

typedef __attribute__((ext_vector_type(16))) _Float16 v16h;
typedef __attribute__((ext_vector_type(8)))  _Float16 v8h;
typedef __attribute__((ext_vector_type(8)))  float    v8f;

#define NB    4096          // batch
#define ND    256           // dim
#define NN    8192          // 2*NB
#define JCHUNKS 8           // column-range splits
#define JTILES_PER_CHUNK (NN / 16 / JCHUNKS)   // 64

// ---------------------------------------------------------------------------
// Fragment loaders for Zn (f16, row-major [NN][ND]) used as WMMA A and B.
//
// A 16x32 f16 layout (ISA 7.12.2): lane L -> row m=L%16;
//   halves 0..7  : K = kw*32 + (L>>4)*8 + h
//   halves 8..15 : K = kw*32 + 16 + (L>>4)*8 + (h-8)
__device__ __forceinline__ v16h load_fragA(const _Float16* zn, int row0, int kw, int lane) {
    int m = lane & 15;
    int g = (lane >> 4) << 3;                      // 0 or 8
    const _Float16* p = zn + (size_t)(row0 + m) * ND + kw * 32 + g;
    v8h lo = *(const v8h*)(p);                     // K offsets 0..7
    v8h hi = *(const v8h*)(p + 16);                // K offsets 16..23
    return __builtin_shufflevector(lo, hi, 0,1,2,3,4,5,6,7,8,9,10,11,12,13,14,15);
}

// B 32x16 f16 layout (per ISA sparse-B pattern scaled to K=32): lane L -> col n=L%16;
//   halves 0..7 : K = kw*32 + (L>>4)*8 + h ; halves 8..15 : K = kw*32 + 16 + (L>>4)*8 + (h-8)
// B column n of (Zn Zn^T) tile is Zn row (col0+n), so load straight from Zn.
__device__ __forceinline__ v16h load_fragB(const _Float16* zn, int col0, int kw, int lane) {
    int n = lane & 15;
    int g = (lane >> 4) << 3;                      // 0 or 8
    const _Float16* p = zn + (size_t)(col0 + n) * ND + kw * 32 + g;
    v8h lo = *(const v8h*)(p);
    v8h hi = *(const v8h*)(p + 16);
    return __builtin_shufflevector(lo, hi, 0,1,2,3,4,5,6,7,8,9,10,11,12,13,14,15);
}

// ---------------------------------------------------------------------------
// Kernel 1: per-row L2-normalize (fp32 math), emit Zn in f16, and exact fp32
// pos[row] = dot(zn_row, zn_partner)/TEMP. One wave32 per row.
__global__ void ntxent_norm(const float* __restrict__ zi, const float* __restrict__ zj,
                            _Float16* __restrict__ zn, float* __restrict__ pos) {
    int lane = threadIdx.x & 31;
    int row  = blockIdx.x * (blockDim.x >> 5) + (threadIdx.x >> 5);
    if (row >= NN) return;

    const float* zr = (row < NB) ? (zi + (size_t)row * ND) : (zj + (size_t)(row - NB) * ND);
    const float* zp = (row < NB) ? (zj + (size_t)row * ND) : (zi + (size_t)(row - NB) * ND);

    float4 a0 = ((const float4*)zr)[lane * 2 + 0];
    float4 a1 = ((const float4*)zr)[lane * 2 + 1];
    float4 b0 = ((const float4*)zp)[lane * 2 + 0];
    float4 b1 = ((const float4*)zp)[lane * 2 + 1];

    float ss = a0.x*a0.x + a0.y*a0.y + a0.z*a0.z + a0.w*a0.w
             + a1.x*a1.x + a1.y*a1.y + a1.z*a1.z + a1.w*a1.w;
    float sp = b0.x*b0.x + b0.y*b0.y + b0.z*b0.z + b0.w*b0.w
             + b1.x*b1.x + b1.y*b1.y + b1.z*b1.z + b1.w*b1.w;
    float dp = a0.x*b0.x + a0.y*b0.y + a0.z*b0.z + a0.w*b0.w
             + a1.x*b1.x + a1.y*b1.y + a1.z*b1.z + a1.w*b1.w;

    #pragma unroll
    for (int off = 1; off < 32; off <<= 1) {
        ss += __shfl_xor(ss, off, 32);
        sp += __shfl_xor(sp, off, 32);
        dp += __shfl_xor(dp, off, 32);
    }

    float invr = 1.0f / fmaxf(sqrtf(ss), 1e-8f);
    float invp = 1.0f / fmaxf(sqrtf(sp), 1e-8f);

    v8h h;
    h[0] = (_Float16)(a0.x * invr); h[1] = (_Float16)(a0.y * invr);
    h[2] = (_Float16)(a0.z * invr); h[3] = (_Float16)(a0.w * invr);
    h[4] = (_Float16)(a1.x * invr); h[5] = (_Float16)(a1.y * invr);
    h[6] = (_Float16)(a1.z * invr); h[7] = (_Float16)(a1.w * invr);
    *(v8h*)(zn + (size_t)row * ND + lane * 8) = h;

    if (lane == 0) pos[row] = dp * invr * invp * 2.0f;   // /TEMP, TEMP=0.5
}

// ---------------------------------------------------------------------------
// Kernel 2: fused GEMM + online exp-sum.
// Each wave owns TWO 16-row M-tiles (A resident, 128 VGPRs) so every B
// fragment feeds two WMMAs (16 WMMA : 16 b128-loads per j-tile).
// __launch_bounds__(256, 1): allow the full per-wave VGPR budget so the 16
// resident A fragments are NOT spilled to scratch (round-2 regression).
// All 8 waves of a block share the SAME j-chunk -> identical B stream,
// WGP$-resident. Block b: chunk = b>>5, waves cover 8 adjacent 32-row blocks.
// partS[chunk][row] = sum over the chunk's j-range of exp(sim(row,j) - 2).
__global__ void __launch_bounds__(256, 1)
ntxent_sim_lse(const _Float16* __restrict__ zn, float* __restrict__ partS) {
    int lane  = threadIdx.x & 31;
    int w     = threadIdx.x >> 5;                  // wave in block, 0..7
    int chunk = blockIdx.x >> 5;                   // 0..7 (shared by whole block)
    int mpair = (blockIdx.x & 31) * 8 + w;         // 0..255
    int row0  = mpair * 32;                        // two 16-row tiles: row0, row0+16

    v16h a0[8], a1[8];
    #pragma unroll
    for (int kw = 0; kw < 8; ++kw) {
        a0[kw] = load_fragA(zn, row0,      kw, lane);
        a1[kw] = load_fragA(zn, row0 + 16, kw, lane);
    }

    float acc0[8], acc1[8];
    #pragma unroll
    for (int r = 0; r < 8; ++r) { acc0[r] = 0.0f; acc1[r] = 0.0f; }

    const int j0 = chunk * (JTILES_PER_CHUNK * 16);
    // per-lane base pointer into the B stream (col = lane&15, halves at g)
    const _Float16* bbase = zn + (size_t)(j0 + (lane & 15)) * ND + ((lane >> 4) << 4);

    #pragma unroll 1
    for (int jt = 0; jt < JTILES_PER_CHUNK; ++jt) {
        int j = j0 + jt * 16;
        // prefetch next tile's B rows into L1 while this tile computes
        if (jt + 1 < JTILES_PER_CHUNK)
            __builtin_prefetch(bbase + (size_t)(jt + 1) * 16 * ND, 0, 3);

        v8f c0 = {}, c1 = {};
        #pragma unroll
        for (int kw = 0; kw < 8; ++kw) {
            v16h b = load_fragB(zn, j, kw, lane);
            c0 = __builtin_amdgcn_wmma_f32_16x16x32_f16(
                     false, a0[kw], false, b, (short)0, c0, false, false);
            c1 = __builtin_amdgcn_wmma_f32_16x16x32_f16(
                     false, a1[kw], false, b, (short)0, c1, false, false);
        }
        // sim = dot/TEMP = 2*c ; accumulate exp(sim - 2) (exact LSE shift, diag max = 2)
        #pragma unroll
        for (int r = 0; r < 8; ++r) {
            acc0[r] += __expf(2.0f * c0[r] - 2.0f);
            acc1[r] += __expf(2.0f * c1[r] - 2.0f);
        }
    }

    // C layout: row m = r + 8*(lane>>4), col = lane&15 -> reduce over 16-lane groups
    #pragma unroll
    for (int r = 0; r < 8; ++r) {
        float v = acc0[r];
        #pragma unroll
        for (int off = 1; off < 16; off <<= 1) v += __shfl_xor(v, off, 32);
        acc0[r] = v;
        float u = acc1[r];
        #pragma unroll
        for (int off = 1; off < 16; off <<= 1) u += __shfl_xor(u, off, 32);
        acc1[r] = u;
    }

    if ((lane & 15) == 0) {
        int g = (lane >> 4) << 3;    // 0 or 8
        float* dst = partS + (size_t)chunk * NN + row0 + g;
        #pragma unroll
        for (int r = 0; r < 8; ++r) dst[r]      = acc0[r];
        #pragma unroll
        for (int r = 0; r < 8; ++r) dst[16 + r] = acc1[r];
    }
}

// ---------------------------------------------------------------------------
// Kernel 3: deterministic final reduction. loss = sum_row(2 + log(S) - pos)/N
__global__ void ntxent_finalize(const float* __restrict__ partS, const float* __restrict__ pos,
                                float* __restrict__ out) {
    __shared__ float red[256];
    int t = threadIdx.x;
    float local = 0.0f;
    for (int row = t; row < NN; row += 256) {
        float S = 0.0f;
        #pragma unroll
        for (int c = 0; c < JCHUNKS; ++c) S += partS[(size_t)c * NN + row];
        local += 2.0f + __logf(S) - pos[row];
    }
    red[t] = local;
    __syncthreads();
    for (int s = 128; s > 0; s >>= 1) {
        if (t < s) red[t] += red[t + s];
        __syncthreads();
    }
    if (t == 0) out[0] = red[0] / (float)NN;
}

// ---------------------------------------------------------------------------
extern "C" void kernel_launch(void* const* d_in, const int* in_sizes, int n_in,
                              void* d_out, int out_size, void* d_ws, size_t ws_size,
                              hipStream_t stream) {
    const float* zi = (const float*)d_in[0];
    const float* zj = (const float*)d_in[1];
    float* out = (float*)d_out;

    char* ws = (char*)d_ws;
    _Float16* zn  = (_Float16*)(ws);                                   // 4 MB
    float*    pos = (float*)(ws + (size_t)NN * ND * sizeof(_Float16)); // 32 KB
    float*    pS  = (float*)((char*)pos + (size_t)NN * sizeof(float)); // 256 KB

    // 1 wave per row, 8 waves (256 thr) per block
    ntxent_norm<<<NN / 8, 256, 0, stream>>>(zi, zj, zn, pos);

    // 256 blocks x 8 waves; each wave = 32 rows x 1 j-chunk
    ntxent_sim_lse<<<(NN / 32 / 8) * JCHUNKS, 256, 0, stream>>>(zn, pS);

    ntxent_finalize<<<1, 256, 0, stream>>>(pS, pos, out);
}
